// G4_12558484374065
// MI455X (gfx1250) — matrix operand order repaired and checked by reading, not compiled, not run on hardware
//
#include <hip/hip_runtime.h>
#include <hip/hip_bf16.h>
#include <math.h>

typedef _Float16 half8 __attribute__((ext_vector_type(8)));
typedef _Float16 v16h  __attribute__((ext_vector_type(16)));
typedef float    v8f   __attribute__((ext_vector_type(8)));
typedef unsigned int u32x4 __attribute__((ext_vector_type(4)));
typedef int          i32x8 __attribute__((ext_vector_type(8)));
typedef int          i32x4 __attribute__((ext_vector_type(4)));

// ---------------------------------------------------------------------------
// Tiny dense layers of the hyper-network head:  out = (relu?)(A @ W + b)
// ---------------------------------------------------------------------------
__global__ void mlp_kernel(const float* __restrict__ A, const float* __restrict__ W,
                           const float* __restrict__ bias, float* __restrict__ out,
                           int K, int N, int do_relu)
{
    int idx = blockIdx.x * 256 + threadIdx.x;          // over 64*N
    if (idx >= 64 * N) return;
    int i = idx / N, j = idx % N;
    float s = bias[j];
    const float* a = A + (size_t)i * K;
    for (int k = 0; k < K; ++k) s += a[k] * W[(size_t)k * N + j];
    if (do_relu && s < 0.f) s = 0.f;
    out[idx] = s;
}

// ---------------------------------------------------------------------------
// Bilinear 7x7 -> 64x64 (half-pixel centers, edge clamp), channel-last f16.
// ---------------------------------------------------------------------------
__global__ void resize_kernel(const float* __restrict__ h3, _Float16* __restrict__ f0)
{
    int idx = blockIdx.x * 256 + threadIdx.x;          // B*4096*8 = 2M
    int c = idx & 7;
    int p = idx >> 3;
    int b = p >> 12, y = (p >> 6) & 63, x = p & 63;
    const float scale = 7.0f / 64.0f;
    float sy = (y + 0.5f) * scale - 0.5f;
    float sx = (x + 0.5f) * scale - 0.5f;
    int y0 = (int)floorf(sy), x0 = (int)floorf(sx);
    float wy = sy - (float)y0, wx = sx - (float)x0;
    int y1 = y0 + 1, x1 = x0 + 1;
    y0 = min(max(y0, 0), 6); y1 = min(max(y1, 0), 6);
    x0 = min(max(x0, 0), 6); x1 = min(max(x1, 0), 6);
    const float* hb = h3 + ((size_t)b * 8 + c) * 49;
    float v = (1.f - wy) * ((1.f - wx) * hb[y0 * 7 + x0] + wx * hb[y0 * 7 + x1])
            +        wy  * ((1.f - wx) * hb[y1 * 7 + x0] + wx * hb[y1 * 7 + x1]);
    f0[idx] = (_Float16)v;
}

// ---------------------------------------------------------------------------
// Implicit-kernel MLPs -> conv weights in WMMA-B layout kw[cout][K] (f16).
// ---------------------------------------------------------------------------
__device__ __forceinline__ float ik_eval(const float* w1, const float* b1,
                                         const float* w2, const float* b2,
                                         int tap, int col)
{
    const float offs[5] = {-0.125f, -0.0625f, 0.f, 0.0625f, 0.125f};
    float gx = offs[tap / 5], gy = offs[tap % 5];
    float s = b2[col];
    for (int h = 0; h < 64; ++h) {
        float hv = gx * w1[h] + gy * w1[64 + h] + b1[h];
        if (hv > 0.f) s += hv * w2[(size_t)h * 1024 + col];
    }
    return s;
}

__global__ void build_k1_kernel(const float* __restrict__ w1, const float* __restrict__ b1,
                                const float* __restrict__ w2, const float* __restrict__ b2,
                                _Float16* __restrict__ kw)   // (128, 224)
{
    int idx = blockIdx.x * 256 + threadIdx.x;
    if (idx >= 128 * 224) return;
    int o = idx / 224, kk = idx % 224;
    int t = kk >> 3, i = kk & 7;                  // K = tap*8 + cin
    float v = 0.f;
    if (t < 25) v = ik_eval(w1, b1, w2, b2, t, i * 128 + o);
    kw[idx] = (_Float16)v;
}

__global__ void build_k2_kernel(const float* __restrict__ w1, const float* __restrict__ b1,
                                const float* __restrict__ w2, const float* __restrict__ b2,
                                _Float16* __restrict__ kw)   // (16, 3200), rows 8..15 zero
{
    int idx = blockIdx.x * 256 + threadIdx.x;
    if (idx >= 16 * 3200) return;
    int o = idx / 3200, kk = idx % 3200;
    float v = 0.f;
    if (o < 8) {
        int t = kk >> 7, i = kk & 127;            // K = tap*128 + cin
        v = ik_eval(w1, b1, w2, b2, t, i * 8 + o);
    }
    kw[idx] = (_Float16)v;
}

// ---------------------------------------------------------------------------
// TDM stage of the conv1 weight matrix (128 rows x 224 K f16, 56 KB) into LDS.
// 2-D D# per ISA ch.8: group0 = {flags, lds_addr, global_addr, type=2},
// group1 = {data_size=2B, tensor_dim0=224, tensor_dim1=128, tile=224x128,
//           dim0_stride=224}.  Issued by wave 0 only; TENSORcnt waited, then
// workgroup barrier makes the LDS tile visible to all 8 waves.
// This toolchain exposes the 6-arg builtin:
//   (u32x4 g0, i32x8 g1, i32x4 g2, i32x4 g3, i32x8 pad, i32 cpol)
// ---------------------------------------------------------------------------
__device__ __forceinline__ void tdm_stage_k1(const _Float16* kw, _Float16* lds)
{
#if __has_builtin(__builtin_amdgcn_tensor_load_to_lds) && __has_builtin(__builtin_amdgcn_s_wait_tensorcnt)
    if (threadIdx.x < 32) {
        unsigned long long ga = (unsigned long long)(uintptr_t)kw;
        unsigned int ldsoff   = (unsigned int)(uintptr_t)lds;   // low 32b = LDS byte offset
        u32x4 g0;
        g0[0] = 1u;                                    // count=1, user descriptor
        g0[1] = ldsoff;                                // lds_addr
        g0[2] = (unsigned int)ga;                      // global_addr[31:0]
        g0[3] = (unsigned int)((ga >> 32) & 0x01FFFFFFu) | 0x80000000u; // addr[56:32] | type=2
        i32x8 g1;
        g1[0] = 0x00010000;                            // wg_mask=0, data_size=1 (2 bytes)
        g1[1] = (224 << 16);                           // tensor_dim0 = 224 (low16 @ bit48)
        g1[2] = (128 << 16);                           // dim0 hi=0 | tensor_dim1 = 128
        g1[3] = (224 << 16);                           // dim1 hi=0 | tile_dim0 = 224
        g1[4] = 128;                                   // tile_dim1 = 128, tile_dim2 = 0
        g1[5] = 224;                                   // tensor_dim0_stride = 224
        g1[6] = 0;
        g1[7] = 0;
        i32x4 z4 = {0, 0, 0, 0};
        i32x8 z8 = {0, 0, 0, 0, 0, 0, 0, 0};
        __builtin_amdgcn_tensor_load_to_lds(g0, g1, z4, z4, z8, 0);
        __builtin_amdgcn_s_wait_tensorcnt(0);
    }
#else
    // cooperative fallback: plain global->LDS copy (16B per thread per iter)
    for (int i = threadIdx.x; i < 128 * 224 / 8; i += 256)
        *(half8*)(lds + i * 8) = *(const half8*)(kw + i * 8);
#endif
    __syncthreads();
}

// ---------------------------------------------------------------------------
// conv1 as implicit GEMM on v_wmma_f32_16x16x32_f16.
// M = B*4096 px, N = 128, K = 224 (7 k-steps).  One 16-px M-tile per block,
// wave w = N-tile w.  B fragments come from the TDM-staged LDS tile
// (ds_load_b128); A fragments are dilated-tap im2col b128 global loads.
// ---------------------------------------------------------------------------
__global__ void conv1_wmma_kernel(const _Float16* __restrict__ f0,  // (B*4096, 8)
                                  const _Float16* __restrict__ kw,  // (128, 224)
                                  _Float16* __restrict__ out)       // (B*4096, 128)
{
    __shared__ _Float16 bsm[128 * 224];          // 56 KB weight tile
    tdm_stage_k1(kw, bsm);

    const int mtile = blockIdx.x;
    const int wave  = threadIdx.x >> 5;      // N-tile 0..7
    const int lane  = threadIdx.x & 31;
    const int m     = lane & 15;
    const int hi    = lane >> 4;
    const int p     = mtile * 16 + m;
    const int b     = p >> 12;
    const int y     = (p >> 6) & 63;
    const int x     = p & 63;
    const int col   = wave * 16 + (lane & 15);

    v8f acc = {};
    #pragma unroll
    for (int step = 0; step < 7; ++step) {
        v16h a;
        #pragma unroll
        for (int c = 0; c < 2; ++c) {
            int kbase = step * 32 + hi * 8 + c * 16;
            int tap   = kbase >> 3;
            half8 v = {};
            if (tap < 25) {
                int dy = 2 * (tap / 5 - 2);
                int dx = 2 * (tap % 5 - 2);
                int yy = y + dy, xx = x + dx;
                if (yy >= 0 && yy < 64 && xx >= 0 && xx < 64) {
                    v = *(const half8*)(f0 + (((size_t)b << 12) + yy * 64 + xx) * 8);
                }
            }
            #pragma unroll
            for (int i = 0; i < 8; ++i) a[c * 8 + i] = v[i];
        }
        v16h bf;
        {
            const _Float16* bp = bsm + col * 224 + step * 32 + hi * 8;  // LDS
            half8 v0 = *(const half8*)bp;
            half8 v1 = *(const half8*)(bp + 16);
            #pragma unroll
            for (int i = 0; i < 8; ++i) { bf[i] = v0[i]; bf[8 + i] = v1[i]; }
        }
        acc = __builtin_amdgcn_wmma_f32_16x16x32_f16(false, a, false, bf,
                                                     (short)0, acc, false, false);
    }
    #pragma unroll
    for (int r = 0; r < 8; ++r) {
        int pp = mtile * 16 + r + hi * 8;
        out[(size_t)pp * 128 + col] = (_Float16)acc[r];
    }
}

// ---------------------------------------------------------------------------
// conv2 implicit GEMM: M = B*4096, N = 16 (8 real), K = 3200.
// Tap loop (25) x fully-unrolled 4 k-steps -> 4 static WMMAs; OOB predicate
// evaluated once per tap; prefetch of the next tap's activation row.
// ---------------------------------------------------------------------------
__global__ void conv2_wmma_kernel(const _Float16* __restrict__ f1,  // (B*4096, 128)
                                  const _Float16* __restrict__ kw,  // (16, 3200)
                                  _Float16* __restrict__ out)       // (B*4096, 8)
{
    const int wave  = threadIdx.x >> 5;
    const int lane  = threadIdx.x & 31;
    const int mtile = blockIdx.x * 8 + wave;
    const int m     = lane & 15;
    const int hi    = lane >> 4;
    const int p     = mtile * 16 + m;
    const int b     = p >> 12;
    const int y     = (p >> 6) & 63;
    const int x     = p & 63;
    const int col   = lane & 15;

    v8f acc = {};
    for (int tap = 0; tap < 25; ++tap) {
        int dy = 2 * (tap / 5 - 2), dx = 2 * (tap % 5 - 2);
        int yy = y + dy, xx = x + dx;
        bool ok = (yy >= 0 && yy < 64 && xx >= 0 && xx < 64);
        const _Float16* arow =
            f1 + ((((size_t)b << 12) + yy * 64 + xx) * 128);
        if (ok && tap < 24) {
            // prefetch next tap's row (same column, shifted) into cache
            __builtin_prefetch(arow + 2 * 128, 0, 0);
        }
        #pragma unroll
        for (int sub = 0; sub < 4; ++sub) {
            int step = tap * 4 + sub;
            v16h a;
            #pragma unroll
            for (int c = 0; c < 2; ++c) {
                half8 v = {};
                if (ok) v = *(const half8*)(arow + sub * 32 + hi * 8 + c * 16);
                #pragma unroll
                for (int i = 0; i < 8; ++i) a[c * 8 + i] = v[i];
            }
            v16h bf;
            {
                const _Float16* bp = kw + (size_t)col * 3200 + step * 32 + hi * 8;
                half8 v0 = *(const half8*)bp;
                half8 v1 = *(const half8*)(bp + 16);
                #pragma unroll
                for (int i = 0; i < 8; ++i) { bf[i] = v0[i]; bf[8 + i] = v1[i]; }
            }
            acc = __builtin_amdgcn_wmma_f32_16x16x32_f16(false, a, false, bf,
                                                         (short)0, acc, false, false);
        }
    }
    if (col < 8) {
        #pragma unroll
        for (int r = 0; r < 8; ++r) {
            int pp = mtile * 16 + r + hi * 8;
            out[(size_t)pp * 8 + col] = (_Float16)acc[r];
        }
    }
}

// ---------------------------------------------------------------------------
// Instance norm: per-(b, c) mean / rsqrt(var+1e-5) over 4096 spatial elems.
// ---------------------------------------------------------------------------
__global__ void inorm_reduce_kernel(const _Float16* __restrict__ src, int nch,
                                    float* __restrict__ mu, float* __restrict__ rs)
{
    int b = blockIdx.x / nch, c = blockIdx.x % nch;
    const _Float16* base = src + (size_t)b * 4096 * nch + c;
    float s = 0.f, sq = 0.f;
    for (int i = threadIdx.x; i < 4096; i += 256) {
        float v = (float)base[(size_t)i * nch];
        s += v; sq += v * v;
    }
    __shared__ float sh1[256], sh2[256];
    sh1[threadIdx.x] = s; sh2[threadIdx.x] = sq;
    __syncthreads();
    for (int o = 128; o > 0; o >>= 1) {
        if (threadIdx.x < o) {
            sh1[threadIdx.x] += sh1[threadIdx.x + o];
            sh2[threadIdx.x] += sh2[threadIdx.x + o];
        }
        __syncthreads();
    }
    if (threadIdx.x == 0) {
        float mean = sh1[0] * (1.f / 4096.f);
        float var  = sh2[0] * (1.f / 4096.f) - mean * mean;
        mu[blockIdx.x] = mean;
        rs[blockIdx.x] = rsqrtf(var + 1e-5f);
    }
}

__global__ void inorm_apply_kernel(_Float16* __restrict__ f, const float* __restrict__ mu,
                                   const float* __restrict__ rs, const float* __restrict__ g,
                                   const float* __restrict__ be, int nch)
{
    size_t idx = (size_t)blockIdx.x * 256 + threadIdx.x;   // B*4096*nch
    int c = (int)(idx % nch);
    int b = (int)(idx / ((size_t)4096 * nch));
    int bc = b * nch + c;
    float v = (float)f[idx];
    v = (v - mu[bc]) * rs[bc] * g[c] + be[c];
    f[idx] = (_Float16)(v > 0.f ? v : 0.f);
}

// ---------------------------------------------------------------------------
// Fused inorm2 + ReLU + channel mix (8->3) + tanh, NCHW f32 output.
// ---------------------------------------------------------------------------
__global__ void finalize_kernel(const _Float16* __restrict__ c2, const float* __restrict__ mu,
                                const float* __restrict__ rs, const float* __restrict__ g2,
                                const float* __restrict__ be2, const float* __restrict__ Wcm,
                                const float* __restrict__ bcm, float* __restrict__ out)
{
    int p = blockIdx.x * 256 + threadIdx.x;   // B*4096 pixels
    int b = p >> 12, sp = p & 4095;
    float v[8];
    #pragma unroll
    for (int c = 0; c < 8; ++c) {
        float t = (float)c2[(size_t)p * 8 + c];
        int bc = b * 8 + c;
        t = (t - mu[bc]) * rs[bc] * g2[c] + be2[c];
        v[c] = t > 0.f ? t : 0.f;
    }
    #pragma unroll
    for (int o = 0; o < 3; ++o) {
        float s = bcm[o];
        #pragma unroll
        for (int c = 0; c < 8; ++c) s += v[c] * Wcm[c * 3 + o];
        out[((size_t)(b * 3 + o) << 12) + sp] = tanhf(s);
    }
}

// ---------------------------------------------------------------------------
extern "C" void kernel_launch(void* const* d_in, const int* in_sizes, int n_in,
                              void* d_out, int out_size, void* d_ws, size_t ws_size,
                              hipStream_t stream)
{
    const float* x    = (const float*)d_in[0];
    const float* W1   = (const float*)d_in[1];
    const float* b1   = (const float*)d_in[2];
    const float* W2   = (const float*)d_in[3];
    const float* b2   = (const float*)d_in[4];
    const float* W3   = (const float*)d_in[5];
    const float* b3   = (const float*)d_in[6];
    const float* k1w1 = (const float*)d_in[7];
    const float* k1b1 = (const float*)d_in[8];
    const float* k1w2 = (const float*)d_in[9];
    const float* k1b2 = (const float*)d_in[10];
    const float* g1   = (const float*)d_in[11];
    const float* be1  = (const float*)d_in[12];
    const float* k2w1 = (const float*)d_in[13];
    const float* k2b1 = (const float*)d_in[14];
    const float* k2w2 = (const float*)d_in[15];
    const float* k2b2 = (const float*)d_in[16];
    const float* g2   = (const float*)d_in[17];
    const float* be2  = (const float*)d_in[18];
    const float* Wcm  = (const float*)d_in[19];
    const float* bcm  = (const float*)d_in[20];
    float* out = (float*)d_out;

    // workspace carve-up (all chunk sizes are multiples of 256 bytes)
    char* ws = (char*)d_ws;
    size_t off = 0;
    float*    h1   = (float*)(ws + off);    off += 64 * 512 * sizeof(float);
    float*    h2   = (float*)(ws + off);    off += 64 * 256 * sizeof(float);
    float*    h3   = (float*)(ws + off);    off += 64 * 392 * sizeof(float);
    _Float16* kw1  = (_Float16*)(ws + off); off += 128 * 224 * sizeof(_Float16);
    _Float16* kw2  = (_Float16*)(ws + off); off += 16 * 3200 * sizeof(_Float16);
    _Float16* f0   = (_Float16*)(ws + off); off += (size_t)64 * 4096 * 8 * 2;
    _Float16* c1   = (_Float16*)(ws + off); off += (size_t)64 * 4096 * 128 * 2;
    float*    mu1  = (float*)(ws + off);    off += 64 * 128 * sizeof(float);
    float*    rs1  = (float*)(ws + off);    off += 64 * 128 * sizeof(float);
    _Float16* c2   = (_Float16*)(ws + off); off += (size_t)64 * 4096 * 8 * 2;
    float*    mu2  = (float*)(ws + off);    off += 64 * 8 * sizeof(float) + 2048;
    float*    rs2  = (float*)(ws + off);    off += 64 * 8 * sizeof(float) + 2048;
    (void)ws_size; (void)in_sizes; (void)n_in; (void)out_size;

    // 1) hyper-network head:  x -> h1 -> h2 -> h3 (B,8,7,7)
    mlp_kernel<<<(64 * 512 + 255) / 256, 256, 0, stream>>>(x,  W1, b1, h1, 128, 512, 1);
    mlp_kernel<<<(64 * 256 + 255) / 256, 256, 0, stream>>>(h1, W2, b2, h2, 512, 256, 1);
    mlp_kernel<<<(64 * 392 + 255) / 256, 256, 0, stream>>>(h2, W3, b3, h3, 256, 392, 0);

    // 2) bilinear INR evaluation 7x7 -> 64x64, channel-last f16
    resize_kernel<<<(64 * 4096 * 8) / 256, 256, 0, stream>>>(h3, f0);

    // 3) implicit conv kernels in WMMA-B layout
    build_k1_kernel<<<(128 * 224 + 255) / 256, 256, 0, stream>>>(k1w1, k1b1, k1w2, k1b2, kw1);
    build_k2_kernel<<<(16 * 3200 + 255) / 256, 256, 0, stream>>>(k2w1, k2b1, k2w2, k2b2, kw2);

    // 4) conv1 (8->128) implicit GEMM on WMMA, weights staged via TDM
    conv1_wmma_kernel<<<16384, 256, 0, stream>>>(f0, kw1, c1);

    // 5) instance norm 1 + ReLU (in place, f16)
    inorm_reduce_kernel<<<64 * 128, 256, 0, stream>>>(c1, 128, mu1, rs1);
    inorm_apply_kernel<<<(size_t)64 * 4096 * 128 / 256, 256, 0, stream>>>(c1, mu1, rs1, g1, be1, 128);

    // 6) conv2 (128->8) implicit GEMM on WMMA
    conv2_wmma_kernel<<<2048, 256, 0, stream>>>(c1, kw2, c2);

    // 7) instance norm 2 stats, then fused norm+ReLU+mix+tanh to NCHW f32
    inorm_reduce_kernel<<<64 * 8, 256, 0, stream>>>(c2, 8, mu2, rs2);
    finalize_kernel<<<(64 * 4096) / 256, 256, 0, stream>>>(c2, mu2, rs2, g2, be2, Wcm, bcm, out);
}
